// BinarizedLinear_74440373174392
// MI455X (gfx1250) — 8-bit matrix operand order repaired and checked by reading, compiled, not run on hardware
//
#include <hip/hip_runtime.h>
#include <stdint.h>

typedef int v8i __attribute__((ext_vector_type(8)));

#define M_DIM 8192
#define N_DIM 4096
#define K_DIM 4096
#define KT    (K_DIM / 64)
#define MT    (M_DIM / 16)
#define NT    (N_DIM / 16)
#define TILE_BYTES 1024

__device__ __forceinline__ uint32_t pack4signs(float4 v) {
    uint32_t b0 = (v.x + 1e-20f >= 0.0f) ? 0x01u : 0xFFu;
    uint32_t b1 = (v.y + 1e-20f >= 0.0f) ? 0x01u : 0xFFu;
    uint32_t b2 = (v.z + 1e-20f >= 0.0f) ? 0x01u : 0xFFu;
    uint32_t b3 = (v.w + 1e-20f >= 0.0f) ? 0x01u : 0xFFu;
    return b0 | (b1 << 8) | (b2 << 16) | (b3 << 24);
}

__global__ void __launch_bounds__(256)
pack_a_kernel(const float* __restrict__ x, uint8_t* __restrict__ apack) {
    const int tile  = blockIdx.x;
    const int mTile = tile / KT;
    const int kTile = tile % KT;
    const int lane  = threadIdx.x & 31;
    const int j     = threadIdx.x >> 5;

    const int m    = mTile * 16 + (lane & 15);
    const int kOff = (j & 1) * 4 + ((j >> 1) & 1) * 16 + ((j >> 2) & 1) * 32
                   + (lane >> 4) * 8;
    const int k    = kTile * 64 + kOff;

    const float4 f = *(const float4*)(x + (size_t)m * K_DIM + k);
    *(uint32_t*)(apack + (size_t)tile * TILE_BYTES + lane * 32 + j * 4) =
        pack4signs(f);
}

__global__ void __launch_bounds__(256)
pack_b_kernel(const float* __restrict__ w, uint8_t* __restrict__ bpack) {
    const int tile  = blockIdx.x;
    const int nTile = tile / KT;
    const int kTile = tile % KT;
    const int lane  = threadIdx.x & 31;
    const int j     = threadIdx.x >> 5;

    const int n    = nTile * 16 + (lane & 15);
    const int kOff = (j & 1) * 4 + ((j >> 1) & 1) * 16 + ((j >> 2) & 1) * 32
                   + (lane >> 4) * 8;
    const int k    = kTile * 64 + kOff;

    const float4 f = *(const float4*)(w + (size_t)n * K_DIM + k);
    *(uint32_t*)(bpack + (size_t)tile * TILE_BYTES + lane * 32 + j * 4) =
        pack4signs(f);
}

__global__ void __launch_bounds__(32)
bgemm_kernel(const uint8_t* __restrict__ apack,
             const uint8_t* __restrict__ bpack,
             float* __restrict__ out) {
    const int lane = threadIdx.x;
    const int mT0  = blockIdx.y * 4;
    const int nT0  = blockIdx.x * 4;

    v8i acc[4][4] = {};

    const uint8_t* aBase = apack + (size_t)lane * 32;
    const uint8_t* bBase = bpack + (size_t)lane * 32;

    for (int kt = 0; kt < KT; ++kt) {
        v8i afrag[4], bfrag[4];
#pragma unroll
        for (int i = 0; i < 4; ++i) {
            afrag[i] = *(const v8i*)(aBase +
                       ((size_t)(mT0 + i) * KT + kt) * TILE_BYTES);
            bfrag[i] = *(const v8i*)(bBase +
                       ((size_t)(nT0 + i) * KT + kt) * TILE_BYTES);
        }
#pragma unroll
        for (int i = 0; i < 4; ++i) {
#pragma unroll
            for (int jn = 0; jn < 4; ++jn) {
                acc[i][jn] = __builtin_amdgcn_wmma_i32_16x16x64_iu8(
                    true, afrag[i], true, bfrag[jn], acc[i][jn],
                    false, false);
            }
        }
    }

    const int nCol = lane & 15;
    const int mAdd = (lane >> 4) * 8;
#pragma unroll
    for (int i = 0; i < 4; ++i) {
#pragma unroll
        for (int jn = 0; jn < 4; ++jn) {
            const size_t nG = (size_t)(nT0 + jn) * 16 + nCol;
            float* o = out + nG;
#pragma unroll
            for (int r = 0; r < 8; ++r) {
                const size_t mG = (size_t)(mT0 + i) * 16 + r + mAdd;
                o[mG * N_DIM] = (float)acc[i][jn][r];
            }
        }
    }
}

extern "C" void kernel_launch(void* const* d_in, const int* in_sizes, int n_in,
                              void* d_out, int out_size, void* d_ws,
                              size_t ws_size, hipStream_t stream) {
    (void)in_sizes; (void)n_in; (void)out_size; (void)ws_size;

    const float* x = (const float*)d_in[0];
    const float* w = (const float*)d_in[1];
    float* out     = (float*)d_out;

    uint8_t* apack = (uint8_t*)d_ws;
    uint8_t* bpack = apack + (size_t)M_DIM * K_DIM;

    pack_a_kernel<<<MT * KT, 256, 0, stream>>>(x, apack);
    pack_b_kernel<<<NT * KT, 256, 0, stream>>>(w, bpack);

    dim3 grid(N_DIM / 64, M_DIM / 64);
    bgemm_kernel<<<grid, 32, 0, stream>>>(apack, bpack, out);
}
